// DiffusionMoEModel_26276609917524
// MI455X (gfx1250) — compile-verified, matrix operand order checked by reading, not run on hardware
//
#include <hip/hip_runtime.h>
#include <hip/hip_bf16.h>

// ---------------- model dims ----------------
#define DIMC    768
#define NHEADC  12
#define HDC     64
#define NEXPC   8
#define MULTD   3072      // MULT*DIM
#define TOKS    8192      // B*N
#define SEQC    4096
#define VOCM1   64
#define MAXFRQ  64
#define LORAC   32
#define MOEROWS 16384     // TOKS * top2

typedef __attribute__((ext_vector_type(16))) __bf16        v16bf;
typedef __attribute__((ext_vector_type(8)))  float         v8f;
typedef __attribute__((ext_vector_type(4)))  unsigned int  ui4;
typedef __attribute__((ext_vector_type(8)))  int           i8v;
typedef __attribute__((ext_vector_type(4)))  int           i4v;

union Frag { v16bf v; uint4 q[2]; };

__device__ __forceinline__ unsigned short f2bf(float f) {
  unsigned int u = __float_as_uint(f);
  u += 0x7FFFu + ((u >> 16) & 1u);          // round-to-nearest-even bf16
  return (unsigned short)(u >> 16);
}
__device__ __forceinline__ float bf2f(unsigned short h) {
  return __uint_as_float(((unsigned int)h) << 16);
}
__device__ __forceinline__ float gelu_f(float x) {
  return 0.5f * x * (1.0f + tanhf(0.7978845608028654f * (x + 0.044715f * x * x * x)));
}

// ---------------- Tensor Data Mover 2-D tile load (bf16 elements) ----------------
// D# per CDNA5 ISA ch.8: group0 = {count/lds_addr/global_addr/type}, group1 = {flags/dims}.
// Pads LDS destination 4 dwords every 16 dwords => 64B data + 16B pad == LDT row stride.
#if __has_include(<hip/amd_detail/amd_gfx1250_TDM.h>)
#define TDM_ARITY6 1
#endif

__device__ __forceinline__ void tdm_load_2d(unsigned lds_off, const void* gptr,
                                            unsigned tile_rows, unsigned tile_cols,
                                            unsigned rem_rows, unsigned rem_cols,
                                            unsigned row_stride /* elements */)
{
  unsigned long long ga = (unsigned long long)gptr;
  ui4 g0;
  g0[0] = 1u;                                        // count=1, user descriptor
  g0[1] = lds_off;                                   // lds_addr [63:32]
  g0[2] = (unsigned)(ga & 0xFFFFFFFFu);              // global_addr lo
  g0[3] = (unsigned)((ga >> 32) & 0x01FFFFFFu) | (2u << 30);  // addr hi | type=2
  i8v g1;
  // data_size=1 (2B) | pad_enable | pad_interval=3 (16 dwords) | pad_amount=3 (4 dwords)
  g1[0] = (int)(0x00010000u | (1u << 20) | (3u << 22) | (3u << 25));
  g1[1] = (int)((rem_cols & 0xFFFFu) << 16);                           // tensor_dim0 [63:48]
  g1[2] = (int)(((rem_cols >> 16) & 0xFFFFu) | ((rem_rows & 0xFFFFu) << 16)); // dim0 hi | dim1 lo
  g1[3] = (int)(((rem_rows >> 16) & 0xFFFFu) | (tile_cols << 16));     // dim1 hi | tile_dim0
  g1[4] = (int)(tile_rows & 0xFFFFu);                                  // tile_dim1 (tile_dim2=0)
  g1[5] = (int)row_stride;                                             // tensor_dim0_stride lo32
  g1[6] = 0;                                                           // stride hi | dim1_stride lo
  g1[7] = 0;
  i4v z = {0, 0, 0, 0};
#ifdef TDM_ARITY6
  i8v z8 = {0, 0, 0, 0, 0, 0, 0, 0};
  __builtin_amdgcn_tensor_load_to_lds(g0, g1, z, z, z8, 0);
#else
  __builtin_amdgcn_tensor_load_to_lds(g0, g1, z, z, 0);
#endif
}

// ---------------- generic bf16 WMMA GEMM ----------------
// C[m,n] = f(A[row(m),:] @ Wt[n,:] + bias[n]) (+ resid)   (Wt is [N][K] bf16)
// seg != null => expert e = blockIdx.z, rows [seg[e], seg[e+1]), Wt += e*K*N, bias += e*N.
// Dense path: double-buffered TDM pipeline (DMA for step p+1 in flight under WMMAs of step p).
#define BM 128
#define BN 64
#define BK 32
#define LDT 40   // LDS row stride (elems): 64B data + 16B pad (TDM pad config above)

#define FLAG_GELU  1
#define FLAG_OUTBF 2

template <bool GATHER>
__global__ __launch_bounds__(256) void gemm_bf16_kernel(
    const unsigned short* __restrict__ A, const unsigned short* __restrict__ Wt,
    const float* __restrict__ bias, const float* __restrict__ resid,
    void* __restrict__ Cv, int M, int N, int K, int flags,
    const int* __restrict__ row_map, const int* __restrict__ seg)
{
  __shared__ __align__(16) unsigned short As[2][BM][LDT];
  __shared__ __align__(16) unsigned short Bs[2][BN][LDT];

  const int tid = threadIdx.x;
  int mStart = 0, mEnd = M;
  if (seg) {
    const int e = blockIdx.z;
    mStart = seg[e];
    mEnd   = seg[e + 1];
    Wt   += (size_t)e * K * N;
    if (bias) bias += (size_t)e * N;
  }
  const int m0 = mStart + (int)blockIdx.y * BM;
  if (m0 >= mEnd) return;                    // block-uniform exit (EXEC stays full)
  const int n0 = (int)blockIdx.x * BN;

  const int lane = tid & 31, wid = tid >> 5;
  const int lo = lane & 15, hi = lane >> 4;
  const int wm = wid & 3,  wn = wid >> 2;    // 4x2 waves -> 128x64 tile

  v8f acc[2][2];
  for (int i = 0; i < 2; ++i)
    for (int j = 0; j < 2; ++j)
      for (int r = 0; r < 8; ++r) acc[i][j][r] = 0.0f;

  // fragment build + 4 WMMAs from LDS buffer `buf`
  auto compute_tile = [&](int buf) {
    Frag a[2], b[2];
    const int k0 = hi * 8;   // lanes 0-15: K 0-7 & 16-23 ; lanes 16-31: K 8-15 & 24-31
    for (int i = 0; i < 2; ++i) {
      int r = wm * 32 + i * 16 + lo;
      a[i].q[0] = *(const uint4*)&As[buf][r][k0];
      a[i].q[1] = *(const uint4*)&As[buf][r][k0 + 16];
    }
    for (int j = 0; j < 2; ++j) {
      int r = wn * 32 + j * 16 + lo;
      b[j].q[0] = *(const uint4*)&Bs[buf][r][k0];
      b[j].q[1] = *(const uint4*)&Bs[buf][r][k0 + 16];
    }
    for (int i = 0; i < 2; ++i)
      for (int j = 0; j < 2; ++j)
        acc[i][j] = __builtin_amdgcn_wmma_f32_16x16x32_bf16(
            false, a[i].v, false, b[j].v, (short)0, acc[i][j], false, false);
  };

  if (!GATHER) {
    // ---- dense: double-buffered TDM pipeline ----
    const int T = (K + BK - 1) / BK;
    auto issue_pair = [&](int p) {   // wave-uniform; executed by wave 0 only
      const int kc = p * BK;
      const unsigned buf = (unsigned)(p & 1);
      tdm_load_2d((unsigned)(unsigned long long)&As[buf][0][0],
                  A + (size_t)m0 * K + kc, BM, BK,
                  (unsigned)(mEnd - m0), (unsigned)(K - kc), (unsigned)K);
      tdm_load_2d((unsigned)(unsigned long long)&Bs[buf][0][0],
                  Wt + (size_t)n0 * K + kc, BN, BK,
                  (unsigned)(N - n0), (unsigned)(K - kc), (unsigned)K);
    };
    if (wid == 0) issue_pair(0);
    for (int p = 0; p < T; ++p) {
      if (wid == 0) {
        if (p + 1 < T) {
          issue_pair(p + 1);                         // next tile pair in flight
          __builtin_amdgcn_s_wait_tensorcnt(2);      // oldest pair (this step) landed
        } else {
          __builtin_amdgcn_s_wait_tensorcnt(0);      // final tile
        }
      }
      __syncthreads();
      compute_tile(p & 1);
      __syncthreads();                               // buffer p&1 free for step p+2's TDM
    }
  } else {
    // ---- gather (MoE fc1): manual A staging through row_map, B via TDM ----
    for (int kc = 0; kc < K; kc += BK) {
      if (wid == 0)
        tdm_load_2d((unsigned)(unsigned long long)&Bs[0][0][0],
                    Wt + (size_t)n0 * K + kc, BN, BK,
                    (unsigned)(N - n0), (unsigned)(K - kc), (unsigned)K);
      for (int c = tid; c < BM * 4; c += 256) {      // 16B per chunk, row-level guard only
        int row = c >> 2, q = c & 3;
        int gm = m0 + row;
        uint4 val{0, 0, 0, 0};
        if (gm < mEnd) {
          int ar = row_map[gm];
          val = *(const uint4*)(A + (size_t)ar * K + kc + q * 8);
        }
        *(uint4*)&As[0][row][q * 8] = val;
      }
      if (kc + BK < K)
        __builtin_prefetch(A + (size_t)m0 * K + kc + BK, 0, 1);
      if (wid == 0) __builtin_amdgcn_s_wait_tensorcnt(0);
      __syncthreads();
      compute_tile(0);
      __syncthreads();
    }
  }

  // ---- epilogue: bias / gelu / residual, fp32 or bf16 out ----
  float* Cf = (float*)Cv;
  unsigned short* Ch = (unsigned short*)Cv;
  const int act = flags & FLAG_GELU, outbf = flags & FLAG_OUTBF;
  for (int i = 0; i < 2; ++i)
    for (int j = 0; j < 2; ++j) {
      int n = n0 + wn * 32 + j * 16 + lo;
      for (int r = 0; r < 8; ++r) {
        int m = m0 + wm * 32 + i * 16 + hi * 8 + r;   // C layout: VGPR r -> M = r + 8*hi
        if (m < mEnd && n < N) {
          float v = acc[i][j][r];
          if (bias)  v += bias[n];
          if (act)   v = gelu_f(v);
          if (resid) v += resid[(size_t)m * N + n];
          if (outbf) Ch[(size_t)m * N + n] = f2bf(v);
          else       Cf[(size_t)m * N + n] = v;
        }
      }
    }
}

// ---------------- weight convert: fp32 [K][N] -> bf16 [N][K] (per matrix z) ----------------
__global__ __launch_bounds__(256) void wconv_kernel(const float* __restrict__ W,
                                                    unsigned short* __restrict__ Wt,
                                                    int K, int N)
{
  const size_t mat = (size_t)blockIdx.z;
  const float* Wm = W + mat * (size_t)K * N;
  unsigned short* Wo = Wt + mat * (size_t)K * N;
  long long g = (long long)blockIdx.x * 256 + threadIdx.x;
  if (g >= (long long)K * N) return;
  int k = (int)(g / N), n = (int)(g - (long long)k * N);
  Wo[(size_t)n * K + k] = f2bf(Wm[g]);
}

// ---------------- time-fourier + LoRA embedding prep (B=2) ----------------
__global__ void prep_kernel(const float* __restrict__ time_t, const int* __restrict__ pid,
                            const float* __restrict__ time_w, const float* __restrict__ time_b,
                            const float* __restrict__ pert_base, const float* __restrict__ loraA,
                            const float* __restrict__ loraB,
                            float* __restrict__ temb, float* __restrict__ pemb)
{
  for (int idx = threadIdx.x; idx < 2 * DIMC; idx += blockDim.x) {
    int b = idx / DIMC, d = idx % DIMC;
    float t = time_t[b];
    float acc = time_b[d];
    for (int f = 0; f < MAXFRQ; ++f) {
      float ang = t * (float)f * 3.14159265358979323846f;
      acc += sinf(ang) * time_w[(size_t)f * DIMC + d];
      acc += cosf(ang) * time_w[(size_t)(MAXFRQ + f) * DIMC + d];
    }
    temb[idx] = acc;
    int p = pid[b];
    float pe = pert_base[(size_t)p * DIMC + d];
    for (int r = 0; r < LORAC; ++r)
      pe += loraA[(size_t)p * LORAC + r] * loraB[(size_t)r * DIMC + d];
    pemb[idx] = pe;
  }
}

__global__ void embed_kernel(const int* __restrict__ tokens,
                             const float* __restrict__ token_emb,
                             const float* __restrict__ temb, const float* __restrict__ pemb,
                             float* __restrict__ x)
{
  int g = blockIdx.x * 256 + threadIdx.x;      // < TOKS*DIM
  int t = g / DIMC, d = g - t * DIMC;
  int b = t / SEQC;
  x[g] = token_emb[(size_t)tokens[t] * DIMC + d] + temb[b * DIMC + d] + pemb[b * DIMC + d];
}

// ---------------- LayerNorm: fp32 in -> bf16 out (one block per row) ----------------
__global__ __launch_bounds__(256) void ln_kernel(const float* __restrict__ in,
                                                 unsigned short* __restrict__ out,
                                                 const float* __restrict__ s, const float* __restrict__ b)
{
  const int row = blockIdx.x, tid = threadIdx.x;
  const float* xr = in + (size_t)row * DIMC;
  float s1 = 0.f, s2 = 0.f;
  for (int c = tid; c < DIMC; c += 256) { float v = xr[c]; s1 += v; s2 += v * v; }
  __shared__ float r1[256], r2[256];
  r1[tid] = s1; r2[tid] = s2; __syncthreads();
  for (int off = 128; off > 0; off >>= 1) {
    if (tid < off) { r1[tid] += r1[tid + off]; r2[tid] += r2[tid + off]; }
    __syncthreads();
  }
  float mean = r1[0] * (1.0f / DIMC);
  float var  = r2[0] * (1.0f / DIMC) - mean * mean;   // biased, like jnp.var
  float rstd = rsqrtf(var + 1e-5f);
  for (int c = tid; c < DIMC; c += 256)
    out[(size_t)row * DIMC + c] = f2bf((xr[c] - mean) * rstd * s[c] + b[c]);
}

// ---------------- per-token 12x12 head-attention ----------------
// output in [b][head][n][d] order == reference's scrambled transpose+reshape; bf16 out
__global__ __launch_bounds__(256) void attn_kernel(const float* __restrict__ qkv,
                                                   unsigned short* __restrict__ outP)
{
  const int t = blockIdx.x, tid = threadIdx.x;
  const int b = t / SEQC, n = t - b * SEQC;
  __shared__ float sq[NHEADC][HDC], sk[NHEADC][HDC], sv[NHEADC][HDC];
  __shared__ float att[NHEADC][NHEADC];
  const float* row = qkv + (size_t)t * (3 * DIMC);
  for (int i = tid; i < 3 * DIMC; i += 256) {
    int sec = i / DIMC, idx = i - sec * DIMC;
    int h = idx / HDC, d = idx - h * HDC;
    float v = row[i];
    if (sec == 0) sq[h][d] = v; else if (sec == 1) sk[h][d] = v; else sv[h][d] = v;
  }
  __syncthreads();
  for (int idx = tid; idx < NHEADC * NHEADC; idx += 256) {
    int i = idx / NHEADC, j = idx - i * NHEADC;
    float s = 0.f;
    for (int d = 0; d < HDC; ++d) s += sq[i][d] * sk[j][d];
    att[i][j] = s * 0.125f;                    // 1/sqrt(64)
  }
  __syncthreads();
  if (tid < NHEADC) {
    float mx = att[tid][0];
    for (int j = 1; j < NHEADC; ++j) mx = fmaxf(mx, att[tid][j]);
    float ex[NHEADC], sum = 0.f;
    for (int j = 0; j < NHEADC; ++j) { ex[j] = expf(att[tid][j] - mx); sum += ex[j]; }
    float inv = 1.f / sum;
    for (int j = 0; j < NHEADC; ++j) att[tid][j] = ex[j] * inv;
  }
  __syncthreads();
  for (int idx = tid; idx < DIMC; idx += 256) {
    int h = idx / HDC, d = idx - h * HDC;
    float o = 0.f;
    for (int j = 0; j < NHEADC; ++j) o += att[h][j] * sv[j][d];
    outP[((size_t)(b * NHEADC + h) * SEQC + n) * HDC + d] = f2bf(o);
  }
}

// ---------------- MoE routing ----------------
__global__ void zero_counts_kernel(int* c) { if (threadIdx.x < NEXPC) c[threadIdx.x] = 0; }

__global__ __launch_bounds__(256) void gate_kernel(const unsigned short* __restrict__ xn,
                                                   const float* __restrict__ gw,
                                                   int* __restrict__ sel, float* __restrict__ wgt,
                                                   int* __restrict__ counts)
{
  const int t = blockIdx.x, tid = threadIdx.x;
  const int e = tid >> 5, lane = tid & 31;
  __shared__ float logits[NEXPC];
  float p = 0.f;
  for (int k = lane; k < DIMC; k += 32) p += bf2f(xn[(size_t)t * DIMC + k]) * gw[(size_t)k * NEXPC + e];
  for (int off = 16; off > 0; off >>= 1) p += __shfl_down(p, off, 32);
  if (lane == 0) logits[e] = p;
  __syncthreads();
  if (tid == 0) {
    float mx = logits[0];
    for (int i = 1; i < NEXPC; ++i) mx = fmaxf(mx, logits[i]);
    float ex[NEXPC], sum = 0.f;
    for (int i = 0; i < NEXPC; ++i) { ex[i] = expf(logits[i] - mx); sum += ex[i]; }
    float pr[NEXPC];
    for (int i = 0; i < NEXPC; ++i) pr[i] = ex[i] / sum;
    int e1 = 0;
    for (int i = 1; i < NEXPC; ++i) if (pr[i] > pr[e1]) e1 = i;    // first-max, jnp.argmax semantics
    int e2 = (e1 == 0) ? 1 : 0;
    for (int i = 0; i < NEXPC; ++i) if (i != e1 && pr[i] > pr[e2]) e2 = i;
    float t1 = pr[e1], t2 = pr[e2], den = t1 + t2 + 1e-9f;
    sel[2 * t] = e1; sel[2 * t + 1] = e2;
    wgt[2 * t] = t1 / den; wgt[2 * t + 1] = t2 / den;
    atomicAdd(&counts[e1], 1);
    atomicAdd(&counts[e2], 1);
  }
}

__global__ void scan_kernel(const int* __restrict__ counts, int* __restrict__ seg, int* __restrict__ pos)
{
  if (threadIdx.x == 0 && blockIdx.x == 0) {
    int s = 0;
    for (int e = 0; e < NEXPC; ++e) { seg[e] = s; s += counts[e]; pos[e] = 0; }
    seg[NEXPC] = s;   // == MOEROWS
  }
}

__global__ void build_kernel(const int* __restrict__ sel, const int* __restrict__ seg,
                             int* __restrict__ pos, int* __restrict__ row_map, int* __restrict__ slot)
{
  int t = blockIdx.x * 256 + threadIdx.x;
  if (t >= TOKS) return;
  for (int s = 0; s < 2; ++s) {
    int e = sel[2 * t + s];
    int p = atomicAdd(&pos[e], 1);
    int row = seg[e] + p;
    row_map[row] = t;
    slot[2 * t + s] = row;
  }
}

__global__ void combine_kernel(float* __restrict__ x, const float* __restrict__ y,
                               const float* __restrict__ wgt, const int* __restrict__ slot)
{
  int g = blockIdx.x * 256 + threadIdx.x;     // < TOKS*DIM
  int t = g / DIMC, c = g - t * DIMC;
  x[g] += wgt[2 * t]     * y[(size_t)slot[2 * t]     * DIMC + c]
        + wgt[2 * t + 1] * y[(size_t)slot[2 * t + 1] * DIMC + c];
}

// ---------------- host orchestration ----------------
extern "C" void kernel_launch(void* const* d_in, const int* in_sizes, int n_in,
                              void* d_out, int out_size, void* d_ws, size_t ws_size,
                              hipStream_t stream)
{
  (void)in_sizes; (void)n_in; (void)out_size; (void)ws_size;
  auto F = [&](int i) { return (const float*)d_in[i]; };
  const int* tokens = (const int*)d_in[0];
  const float* time_t = F(1);
  const int* pid = (const int*)d_in[2];
  // params: 3 token_emb, 4 time_w, 5 time_b, 6 pert_base, 7 lora_A, 8 lora_B,
  // dense[i] base 9+10i: ln1_s,ln1_b,qkv_w,proj_w,ln2_s,ln2_b,fc1_w,fc1_b,fc2_w,fc2_b
  // moe[i]   base 29+7i: ln_s,ln_b,gate_w,fc1_w,fc1_b,fc2_w,fc2_b
  // 43 lnf_s, 44 lnf_b, 45 head_w, 46 head_b

  // ---- fp32 workspace ----
  float* wsf = (float*)d_ws;
  size_t o = 0;
  float* temb = wsf + o; o += 2 * DIMC;
  float* pemb = wsf + o; o += 2 * DIMC;
  float* x    = wsf + o; o += (size_t)TOKS * DIMC;
  float* qkv  = wsf + o; o += (size_t)TOKS * 3 * DIMC;
  float* ymoe = wsf + o; o += (size_t)MOEROWS * DIMC;
  float* wgt  = wsf + o; o += 2 * TOKS;
  // ---- int workspace ----
  int* wsi     = (int*)(wsf + o); o += (48 + 2 * TOKS + MOEROWS + 2 * TOKS + 63) & ~63ull;
  int* counts  = wsi;               // 8
  int* pos     = wsi + 8;           // 8
  int* seg     = wsi + 16;          // 9 (pad to 48)
  int* sel     = wsi + 48;          // 2*TOKS
  int* row_map = sel + 2 * TOKS;    // MOEROWS
  int* slot    = row_map + MOEROWS; // 2*TOKS
  // ---- bf16 workspace ----
  unsigned short* wsh = (unsigned short*)(wsf + o);
  size_t h = 0;
  unsigned short* xnB   = wsh + h; h += (size_t)TOKS * DIMC;
  unsigned short* attnB = wsh + h; h += (size_t)TOKS * DIMC;
  unsigned short* h1B   = wsh + h; h += (size_t)MOEROWS * MULTD;
  unsigned short* qkvT[2], *projT[2], *fc1T[2], *fc2T[2], *mfc1T[2], *mfc2T[2];
  for (int i = 0; i < 2; ++i) {
    qkvT[i]  = wsh + h; h += (size_t)DIMC * 3 * DIMC;
    projT[i] = wsh + h; h += (size_t)DIMC * DIMC;
    fc1T[i]  = wsh + h; h += (size_t)DIMC * MULTD;
    fc2T[i]  = wsh + h; h += (size_t)MULTD * DIMC;
    mfc1T[i] = wsh + h; h += (size_t)NEXPC * DIMC * MULTD;
    mfc2T[i] = wsh + h; h += (size_t)NEXPC * MULTD * DIMC;
  }
  unsigned short* headT = wsh + h; h += (size_t)DIMC * VOCM1;

  auto wconv = [&](const float* W, unsigned short* Wt, int K, int N, int nmat) {
    dim3 grid((unsigned)(((size_t)K * N + 255) / 256), 1, (unsigned)nmat);
    wconv_kernel<<<grid, 256, 0, stream>>>(W, Wt, K, N);
  };
  auto gemm = [&](const unsigned short* A, const unsigned short* Wt, const float* bias,
                  const float* resid, void* C, int M, int N, int K, int flags,
                  const int* rmap, const int* segp, int nz) {
    dim3 grid((N + BN - 1) / BN, segp ? (MOEROWS / BM) : ((M + BM - 1) / BM), nz);
    if (rmap)
      gemm_bf16_kernel<true ><<<grid, 256, 0, stream>>>(A, Wt, bias, resid, C, M, N, K, flags, rmap, segp);
    else
      gemm_bf16_kernel<false><<<grid, 256, 0, stream>>>(A, Wt, bias, resid, C, M, N, K, flags, nullptr, segp);
  };

  // ---- one-time (per launch) weight transpose+bf16 ----
  for (int i = 0; i < 2; ++i) {
    const int db = 9 + i * 10, mb = 29 + i * 7;
    wconv(F(db + 2), qkvT[i],  DIMC, 3 * DIMC, 1);
    wconv(F(db + 3), projT[i], DIMC, DIMC, 1);
    wconv(F(db + 6), fc1T[i],  DIMC, MULTD, 1);
    wconv(F(db + 8), fc2T[i],  MULTD, DIMC, 1);
    wconv(F(mb + 3), mfc1T[i], DIMC, MULTD, NEXPC);
    wconv(F(mb + 5), mfc2T[i], MULTD, DIMC, NEXPC);
  }
  wconv(F(45), headT, DIMC, VOCM1, 1);

  // ---- embeddings ----
  prep_kernel<<<1, 256, 0, stream>>>(time_t, pid, F(4), F(5), F(6), F(7), F(8), temb, pemb);
  embed_kernel<<<(TOKS * DIMC) / 256, 256, 0, stream>>>(tokens, F(3), temb, pemb, x);

  for (int i = 0; i < 2; ++i) {
    // ---- dense block i ----
    const int db = 9 + i * 10;
    ln_kernel<<<TOKS, 256, 0, stream>>>(x, xnB, F(db + 0), F(db + 1));
    gemm(xnB, qkvT[i], nullptr, nullptr, qkv, TOKS, 3 * DIMC, DIMC, 0, nullptr, nullptr, 1);
    attn_kernel<<<TOKS, 256, 0, stream>>>(qkv, attnB);
    gemm(attnB, projT[i], nullptr, x, x, TOKS, DIMC, DIMC, 0, nullptr, nullptr, 1);
    ln_kernel<<<TOKS, 256, 0, stream>>>(x, xnB, F(db + 4), F(db + 5));
    gemm(xnB, fc1T[i], F(db + 7), nullptr, h1B, TOKS, MULTD, DIMC, FLAG_GELU | FLAG_OUTBF, nullptr, nullptr, 1);
    gemm(h1B, fc2T[i], F(db + 9), x, x, TOKS, DIMC, MULTD, 0, nullptr, nullptr, 1);

    // ---- moe block i (top-2 routing; masked experts contribute exactly 0) ----
    const int mb = 29 + i * 7;
    ln_kernel<<<TOKS, 256, 0, stream>>>(x, xnB, F(mb + 0), F(mb + 1));
    zero_counts_kernel<<<1, 32, 0, stream>>>(counts);
    gate_kernel<<<TOKS, 256, 0, stream>>>(xnB, F(mb + 2), sel, wgt, counts);
    scan_kernel<<<1, 1, 0, stream>>>(counts, seg, pos);
    build_kernel<<<TOKS / 256, 256, 0, stream>>>(sel, seg, pos, row_map, slot);
    gemm(xnB, mfc1T[i], F(mb + 4), nullptr, h1B, MOEROWS, MULTD, DIMC, FLAG_GELU | FLAG_OUTBF, row_map, seg, NEXPC);
    gemm(h1B, mfc2T[i], F(mb + 6), nullptr, ymoe, MOEROWS, DIMC, MULTD, 0, nullptr, seg, NEXPC);
    combine_kernel<<<(TOKS * DIMC) / 256, 256, 0, stream>>>(x, ymoe, wgt, slot);
  }

  // ---- final LN + head ----
  ln_kernel<<<TOKS, 256, 0, stream>>>(x, xnB, F(43), F(44));
  gemm(xnB, headT, F(46), nullptr, (float*)d_out, TOKS, VOCM1, DIMC, 0, nullptr, nullptr, 1);
}
